// MultiCGLayer_13958643712188
// MI455X (gfx1250) — matrix operand order repaired and checked by reading, not compiled, not run on hardware
//
#include <hip/hip_runtime.h>
#include <hip/hip_bf16.h>

// ---------------------------------------------------------------------------
// Equivariant CG tensor-product message passing (2 layers) for MI455X/gfx1250.
// Wave32 tiles of 16 edges; channel-mix via v_wmma_f32_16x16x32_f16 with f32
// accumulate; sparse CG on VALU from LDS; L2-resident gathers + f32 atomics.
// Per-output-irrep groups fully templated -> static unrolled WMMA schedule.
// ---------------------------------------------------------------------------

typedef __attribute__((ext_vector_type(16))) _Float16 v16h;
typedef __attribute__((ext_vector_type(8)))  float    v8f;

#define MULT      8
#define FEAT_DIM  72
#define NCOMB     15

// Combo tables (deg, l_in, l_out), reference enumeration order.
__constant__ int cDEG[NCOMB] = {0,0,0,1,1,1,1,1,1,2,2,2,2,2,2};
__constant__ int cLNI[NCOMB] = {0,1,2,0,1,1,1,2,2,0,1,1,2,2,2};
__constant__ int cLNO[NCOMB] = {0,1,2,1,0,1,2,1,2,2,1,2,0,1,2};
// Combos grouped by output irrep no (l_out = 0,1,2)
__constant__ int cGCOMB[3][6] = {{0,4,12,0,0,0},{1,3,5,7,10,13},{2,6,8,9,11,14}};
__constant__ int cXOFF[3]   = {0,8,32};  // input block offsets by l
__constant__ int cSHOFF[3]  = {0,1,4};   // sh offsets by degree

struct Ent { int ij; float v; };         // ij = i | (j<<8)
struct CGTab {
    int entOff[NCOMB][6];                // per-combo, per-k entry ranges
    int qmeta[3][48];                    // cig | shoff<<4 | xoff<<8
    Ent ents[NCOMB][128];
};

// ---------------- device CG coefficient generation (exact port) -------------

__device__ inline double dfact(int n) {
    double r = 1.0;
    for (int i = 2; i <= n; ++i) r *= (double)i;
    return r;
}

__device__ double dsu2cg(int j1,int m1,int j2,int m2,int j3,int m3) {
    if (m3 != m1 + m2) return 0.0;
    int vmin = max(max(-j1 + j2 + m3, -j1 + m1), 0);
    int vmax = min(min(j2 + j3 + m1, j3 - j1 + j2), j3 + m3);
    if (vmax < vmin) return 0.0;
    double C = (2.0*j3 + 1.0) *
        (dfact(j3+j1-j2)*dfact(j3-j1+j2)*dfact(j1+j2-j3)*dfact(j3+m3)*dfact(j3-m3)) /
        (dfact(j1+j2+j3+1)*dfact(j1-m1)*dfact(j1+m1)*dfact(j2-m2)*dfact(j2+m2));
    double S = 0.0;
    for (int v = vmin; v <= vmax; ++v) {
        double t = (dfact(j2+j3+m1-v)*dfact(j1-m1+v)) /
                   (dfact(v)*dfact(j3-j1+j2-v)*dfact(j3+m3-v)*dfact(v+j1-j2-m3));
        S += ((v + j2 + m2) & 1) ? -t : t;
    }
    return sqrt(C) * S;
}

struct cplx { double re, im; };
__device__ inline cplx cmk(double r, double i) { cplx c; c.re=r; c.im=i; return c; }
__device__ inline cplx cadd(cplx a, cplx b) { return cmk(a.re+b.re, a.im+b.im); }
__device__ inline cplx cmul(cplx a, cplx b) { return cmk(a.re*b.re-a.im*b.im, a.re*b.im+a.im*b.re); }
__device__ inline cplx cconj(cplx a) { return cmk(a.re, -a.im); }
__device__ inline cplx cscale(cplx a, double s) { return cmk(a.re*s, a.im*s); }

__device__ void r2cmat(int l, cplx q[5][5]) {
    for (int a = 0; a < 5; ++a) for (int b = 0; b < 5; ++b) q[a][b] = cmk(0,0);
    const double is2 = 0.70710678118654752440;
    for (int m = -l; m < 0; ++m) { q[l+m][l-m] = cmk(is2,0); q[l+m][l+m] = cmk(0,-is2); }
    q[l][l] = cmk(1,0);
    for (int m = 1; m <= l; ++m) {
        double s = (m & 1) ? -1.0 : 1.0;
        q[l+m][l+m] = cmk(s*is2, 0); q[l+m][l-m] = cmk(0, s*is2);
    }
    cplx ph;          // (-i)^l
    switch (l & 3) {
        case 0: ph = cmk(1,0);  break;
        case 1: ph = cmk(0,-1); break;
        case 2: ph = cmk(-1,0); break;
        default: ph = cmk(0,1); break;
    }
    for (int a = 0; a <= 2*l; ++a)
        for (int b = 0; b <= 2*l; ++b) q[a][b] = cmul(q[a][b], ph);
}

__device__ void so3cg(int l1, int l2, int l3, float cg[5][5][5]) {
    cplx Q1[5][5], Q2[5][5], Q3[5][5];
    r2cmat(l1, Q1); r2cmat(l2, Q2); r2cmat(l3, Q3);
    for (int j = 0; j <= 2*l1; ++j)
        for (int l = 0; l <= 2*l2; ++l)
            for (int m = 0; m <= 2*l3; ++m) {
                cplx s = cmk(0,0);
                for (int i = 0; i <= 2*l1; ++i)
                    for (int k = 0; k <= 2*l2; ++k)
                        for (int n = 0; n <= 2*l3; ++n) {
                            double c = dsu2cg(l1, i-l1, l2, k-l2, l3, n-l3);
                            if (c != 0.0)
                                s = cadd(s, cscale(cmul(cmul(Q1[i][j], Q2[k][l]),
                                                        cconj(Q3[n][m])), c));
                        }
                cg[j][l][m] = (float)s.re;
            }
}

__global__ void build_cgtab_kernel(CGTab* __restrict__ tab) {
    int tid = threadIdx.x;
    if (tid < NCOMB) {
        int deg = cDEG[tid], li = cLNI[tid], lo = cLNO[tid];
        float cg[5][5][5];
        so3cg(deg, li, lo, cg);
        int cnt = 0, nk = 2*lo + 1;
        for (int k = 0; k < nk; ++k) {
            tab->entOff[tid][k] = cnt;
            for (int i = 0; i <= 2*deg; ++i)
                for (int j = 0; j <= 2*li; ++j) {
                    float v = cg[i][j][k];
                    if (fabsf(v) > 1e-7f && cnt < 128) {
                        tab->ents[tid][cnt].ij = i | (j << 8);
                        tab->ents[tid][cnt].v  = v;
                        ++cnt;
                    }
                }
        }
        for (int k = nk; k <= 5; ++k) tab->entOff[tid][k] = cnt;
    }
    if (tid < 3) {
        int g = tid;
        for (int q = 0; q < 48; ++q) {
            int cig = cGCOMB[g][q >> 3], c = q & 7;
            int li = cLNI[cig], deg = cDEG[cig];
            int xoff = cXOFF[li] + c * (2*li + 1);
            tab->qmeta[g][q] = cig | (cSHOFF[deg] << 4) | (xoff << 8);
        }
    }
}

// ------------- B fragments: weights, padded, in WMMA f16 32x16 B layout -----
// lane L holds B[K = 16*(L/16)+h][N = L%16], h = 0..15.
__global__ void build_bfrag_kernel(const float* __restrict__ wts,
                                   _Float16* __restrict__ bfrag) {
    int fi = blockIdx.x;                 // 0..9 = (layer 0..1) x (5 frags)
    int layer = fi / 5, r = fi % 5;
    int g  = (r == 0) ? 0 : (r <= 2 ? 1 : 2);
    int qb = (r == 0) ? 0 : (r <= 2 ? r - 1 : r - 3);
    int lane = threadIdx.x;              // 32 threads
    int n = lane & 15, krow = lane >> 4;
    int nc = (g == 0) ? 3 : 6;
    for (int h = 0; h < 16; ++h) {
        int q = qb * 32 + krow * 16 + h;
        float v = 0.f;
        if (n < MULT && q < nc * MULT) {
            int cig = cGCOMB[g][q >> 3], c = q & 7;
            v = wts[((layer * NCOMB + cig) * MULT + c) * MULT + n];  // W[ci][c][o]
        }
        bfrag[(size_t)fi * 512 + lane * 16 + h] = (_Float16)v;
    }
}

__global__ void zero_kernel(float* __restrict__ p, int n) {
    int i = blockIdx.x * blockDim.x + threadIdx.x;
    int stride = gridDim.x * blockDim.x;
    for (; i < n; i += stride) p[i] = 0.f;
}

// -------------------------- main layer kernel -------------------------------
// Block = 128 threads = 4 waves; each wave owns a 16-edge tile.
// Per-group work fully specialized at compile time.

template <int G, int NC, int NK, int QB, int FRAGBASE, int OUTBASE>
__device__ __forceinline__
void process_group(int lane, int w,
                   float (&xs)[4][16][FEAT_DIM],
                   float (&shl)[4][16][12],
                   _Float16 (&frag)[4][10][512],
                   int (&tgtl)[4][16],
                   const CGTab* __restrict__ tab,
                   const _Float16* __restrict__ bfrag,
                   float* __restrict__ out) {
    // ---- zero this group's A-fragment region (covers K padding)
    {
        uint4 z = {0u,0u,0u,0u};
        uint4* fp = (uint4*)&frag[w][0][0];
#pragma unroll
        for (int i = 0; i < NK * QB * 2; ++i) fp[i * 32 + lane] = z;
    }
    __syncthreads();

    // ---- sparse CG contraction -> f16 A fragments (16x32, lane L holds k = 16*(L/16)+h, as B)
    {
        constexpr int NITER = NC * MULT * 16 / 32;   // 12 or 24 per lane
#pragma unroll 2
        for (int it = 0; it < NITER; ++it) {
            int t = it * 32 + lane;
            int e = t & 15;
            int q = t >> 4;
            int mv  = tab->qmeta[G][q];
            int cig = mv & 15;
            const float* shp = &shl[w][e][(mv >> 4) & 15];
            const float* xp  = &xs[w][e][(mv >> 8) & 255];
            int qb = q >> 5, ql = q & 31;
            int dlane = e + ((ql >> 4) << 4);
            int h     = ql & 15;
#pragma unroll
            for (int k = 0; k < NK; ++k) {
                float acc = 0.f;
                int t1 = tab->entOff[cig][k + 1];
                for (int tt = tab->entOff[cig][k]; tt < t1; ++tt) {
                    Ent en = tab->ents[cig][tt];
                    acc = fmaf(en.v * shp[en.ij & 255], xp[(en.ij >> 8) & 255], acc);
                }
                frag[w][k * QB + qb][dlane * 16 + h] = (_Float16)acc;
            }
        }
    }
    __syncthreads();

    // ---- WMMA channel mix: out[e,o] = sum_q base[e,q] * W[q,o], K-chained
    const _Float16* bl = bfrag + FRAGBASE * 512;
    v16h b0 = *(const v16h*)(bl + lane * 16);
    v16h b1 = (QB > 1) ? *(const v16h*)(bl + 512 + lane * 16) : b0;

    int o = lane & 15;
    int ebase = (lane >> 4) << 3;

#pragma unroll
    for (int k = 0; k < NK; ++k) {
        v8f acc = {0.f,0.f,0.f,0.f,0.f,0.f,0.f,0.f};
        {
            v16h a0 = *(const v16h*)&frag[w][k * QB + 0][lane * 16];
            acc = __builtin_amdgcn_wmma_f32_16x16x32_f16(
                      false, a0, false, b0, (short)0, acc, false, false);
        }
        if (QB > 1) {
            v16h a1 = *(const v16h*)&frag[w][k * QB + 1][lane * 16];
            acc = __builtin_amdgcn_wmma_f32_16x16x32_f16(
                      false, a1, false, b1, (short)0, acc, false, false);
        }
        // D layout: VGPR r, lane L -> M = r + 8*(L/16), N = L%16
        if (o < MULT) {
#pragma unroll
            for (int r = 0; r < 8; ++r) {
                int node = tgtl[w][ebase + r];
                atomicAdd(out + (size_t)node * FEAT_DIM + OUTBASE + o * NK + k,
                          acc[r]);
            }
        }
    }
    __syncthreads();
}

__global__ __launch_bounds__(128)
void cg_layer_kernel(const float* __restrict__ xin,
                     const int*   __restrict__ eidx,   // [2,E] flat: src then tgt
                     const float* __restrict__ sh0,
                     const float* __restrict__ sh1,
                     const float* __restrict__ sh2,
                     const CGTab* __restrict__ tab,
                     const _Float16* __restrict__ bfrag,  // this layer's 5 frags
                     float* __restrict__ out,
                     int nEdges) {
    __shared__ __align__(16) float    xs[4][16][FEAT_DIM];
    __shared__                float    shl[4][16][12];
    __shared__ __align__(32) _Float16 frag[4][10][512];   // A fragments (k*QB+qb)
    __shared__                int      tgtl[4][16];

    const int lane = threadIdx.x & 31;
    const int w    = threadIdx.x >> 5;
    const int e0   = (blockIdx.x * 4 + w) * 16;

    // ---- stage sh + target ids (lanes 0..15 = edges)
    if (lane < 16) {
        int ed = e0 + lane;
        tgtl[w][lane]   = eidx[nEdges + ed];
        shl[w][lane][0] = sh0[ed];
        shl[w][lane][1] = sh1[ed*3 + 0];
        shl[w][lane][2] = sh1[ed*3 + 1];
        shl[w][lane][3] = sh1[ed*3 + 2];
        shl[w][lane][4] = sh2[ed*5 + 0];
        shl[w][lane][5] = sh2[ed*5 + 1];
        shl[w][lane][6] = sh2[ed*5 + 2];
        shl[w][lane][7] = sh2[ed*5 + 3];
        shl[w][lane][8] = sh2[ed*5 + 4];
    }
    // ---- gather x[src]: 2 lanes per edge, 9 float4 each (L2-resident)
    {
        int e = lane >> 1, half = lane & 1;
        int sidx = eidx[e0 + e];
        const float4* xsrc = (const float4*)(xin + (size_t)sidx * FEAT_DIM + half * 36);
        float4* xdst = (float4*)&xs[w][e][half * 36];
#pragma unroll
        for (int r = 0; r < 9; ++r) xdst[r] = xsrc[r];
    }
    __syncthreads();

    //                G  NC NK QB FB  OUT
    process_group<0, 3, 1, 1, 0,  0>(lane, w, xs, shl, frag, tgtl, tab, bfrag, out);
    process_group<1, 6, 3, 2, 1,  8>(lane, w, xs, shl, frag, tgtl, tab, bfrag, out);
    process_group<2, 6, 5, 2, 3, 32>(lane, w, xs, shl, frag, tgtl, tab, bfrag, out);
}

// --------------------------------- launch -----------------------------------

extern "C" void kernel_launch(void* const* d_in, const int* in_sizes, int n_in,
                              void* d_out, int out_size, void* d_ws, size_t ws_size,
                              hipStream_t stream) {
    const float* x0   = (const float*)d_in[0];
    const int*   eidx = (const int*)  d_in[1];
    const float* sh0  = (const float*)d_in[2];
    const float* sh1  = (const float*)d_in[3];
    const float* sh2  = (const float*)d_in[4];
    const float* wts  = (const float*)d_in[5];

    const int nEdges = in_sizes[2];             // sh0 is [E,1]
    const int nNodes = in_sizes[0] / FEAT_DIM;
    const int nFeat  = nNodes * FEAT_DIM;

    char* ws = (char*)d_ws;
    CGTab*    tab   = (CGTab*)ws;                        // ~16 KB
    _Float16* bfrag = (_Float16*)(ws + 16384);           // 10 KB
    float*    out1  = (float*)(ws + 32768);              // 14.4 MB layer-1 output
    float*    outF  = (float*)d_out;

    build_cgtab_kernel<<<1, 32, 0, stream>>>(tab);
    build_bfrag_kernel<<<10, 32, 0, stream>>>(wts, bfrag);
    zero_kernel<<<1024, 256, 0, stream>>>(out1, nFeat);
    zero_kernel<<<1024, 256, 0, stream>>>(outF, nFeat);

    int nBlocks = nEdges / 64;   // E = 1e6 divisible by 64 (4 waves x 16 edges)
    cg_layer_kernel<<<nBlocks, 128, 0, stream>>>(
        x0, eidx, sh0, sh1, sh2, tab, bfrag, out1, nEdges);
    cg_layer_kernel<<<nBlocks, 128, 0, stream>>>(
        out1, eidx, sh0, sh1, sh2, tab, bfrag + 5 * 512, outF, nEdges);
}
